// DensityGridNN_84713934946395
// MI455X (gfx1250) — matrix operand order repaired and checked by reading, not compiled, not run on hardware
//
#include <hip/hip_runtime.h>

// ---- CDNA5 WMMA types ----
typedef _Float16 h16;
typedef __attribute__((ext_vector_type(16))) _Float16 v16h;
typedef __attribute__((ext_vector_type(8)))  _Float16 v8h;
typedef __attribute__((ext_vector_type(8)))  float    v8f;

// address-space-qualified pointer types for the async global->LDS builtin
typedef __attribute__((address_space(1))) int gas_int;
typedef __attribute__((address_space(3))) int las_int;

#define EMB   128
#define TOT   128
#define BATCH 4
#define NATOM 32
#define LBOX  16.0f

// =============================== MPNN ===============================

__global__ void k_prep_h(const float* __restrict__ pos, const float* __restrict__ embed,
                         const float* __restrict__ wpos, const float* __restrict__ bpos,
                         const int* __restrict__ classes, float* __restrict__ h) {
    int idx = blockIdx.x * blockDim.x + threadIdx.x;           // TOT*EMB
    if (idx >= TOT * EMB) return;
    int i = idx >> 7, e = idx & 127;
    float v = embed[classes[i] * EMB + e] + bpos[e];
    v += pos[i * 3 + 0] * wpos[0 * EMB + e]
       + pos[i * 3 + 1] * wpos[1 * EMB + e]
       + pos[i * 3 + 2] * wpos[2 * EMB + e];
    h[idx] = v;
}

__global__ void k_msg(const float* __restrict__ h, const float* __restrict__ wmsg,
                      const int* __restrict__ edges, int nedge, float* __restrict__ agg) {
    __shared__ float hs[EMB];
    int e = blockIdx.x, o = threadIdx.x;
    int src = edges[e], dst = edges[nedge + e];
#if defined(__gfx1250__) && __has_builtin(__builtin_amdgcn_global_load_async_to_lds_b32)
    // CDNA5 async global->LDS copy (ASYNCcnt domain): stage gathered row in LDS
    // without a VGPR round-trip.
    __builtin_amdgcn_global_load_async_to_lds_b32(
        (gas_int*)(h + src * EMB + o),
        (las_int*)&hs[o], 0, 0);
#if __has_builtin(__builtin_amdgcn_s_wait_asynccnt)
    __builtin_amdgcn_s_wait_asynccnt(0);
#else
    asm volatile("s_wait_asynccnt 0" ::: "memory");
#endif
#else
    hs[o] = h[src * EMB + o];
#endif
    __syncthreads();
    float acc = 0.f;
#pragma unroll 8
    for (int k = 0; k < EMB; ++k) acc += hs[k] * wmsg[k * EMB + o];
    acc = fmaxf(acc, 0.f);
    atomicAdd(&agg[dst * EMB + o], acc);
}

// h += agg @ wupd   (128x128x128 GEMM via v_wmma_f32_16x16x32_f16)
__global__ void k_update_wmma(const float* __restrict__ agg, const float* __restrict__ wupd,
                              float* __restrict__ h) {
    int lane = threadIdx.x & 31;
    int wid  = blockIdx.x * (blockDim.x >> 5) + (threadIdx.x >> 5);  // 64 waves total
    int mi = wid >> 3, ni = wid & 7;
    int hf = lane >> 4, l16 = lane & 15;
    v8f acc = {};
    for (int kc = 0; kc < EMB; kc += 32) {
        v16h a, b;
#pragma unroll
        for (int e2 = 0; e2 < 16; ++e2) {
            int ka = kc + ((e2 >> 3) << 4) + (hf << 3) + (e2 & 7);   // A layout (16-bit, 16x32)
            a[e2] = (h16)agg[(mi * 16 + l16) * EMB + ka];
            int kb = kc + ((e2 >> 3) << 4) + (hf << 3) + (e2 & 7);   // B layout (32x16)
            b[e2] = (h16)wupd[kb * EMB + ni * 16 + l16];
        }
        acc = __builtin_amdgcn_wmma_f32_16x16x32_f16(false, a, false, b, (short)0, acc, false, false);
    }
#pragma unroll
    for (int v = 0; v < 8; ++v) {
        int m = mi * 16 + v + (hf << 3);                             // D layout
        h[m * EMB + ni * 16 + l16] += acc[v];
    }
}

// =========================== feature prep ===========================

__global__ void k_feat(const float* __restrict__ h, const float* __restrict__ pw,
                       const float* __restrict__ pb, int PC, float* __restrict__ feat) {
    int idx = blockIdx.x * blockDim.x + threadIdx.x;           // TOT*PC
    if (idx >= TOT * PC) return;
    int i = idx / PC, pc = idx - i * PC;
    float acc = pb[pc];
    for (int e = 0; e < EMB; ++e) acc += h[i * EMB + e] * pw[e * PC + pc];
    feat[idx] = acc;
}

// fold 1x1x1 proj_conv into per-atom features: feat2[i,o] = sum_pc feat[i,pc]*cw[o,pc]
__global__ void k_feat2(const float* __restrict__ feat, const float* __restrict__ cw,
                        int PC, int C, float* __restrict__ feat2) {
    int idx = blockIdx.x * blockDim.x + threadIdx.x;           // TOT*C
    if (idx >= TOT * C) return;
    int i = idx / C, o = idx - i * C;
    float acc = 0.f;
    for (int pc = 0; pc < PC; ++pc) acc += feat[i * PC + pc] * cw[o * PC + pc];
    feat2[idx] = acc;
}

// weights [O,I,3,3,3] f32 -> f16 pre-swizzled into the B-fragment order k_conv3 uses:
// wS[(chunk*octiles + oc)*512 + lane*16 + e], k = chunk*32 + (lane>=16)*16 + e,
// o = oc*16 + lane%16, flattened k = tap*C + ci; rows >= 27C zeroed.
__global__ void k_wswz(const float* __restrict__ w, int C, int KP, int octiles,
                       h16* __restrict__ wS) {
    int idx = blockIdx.x * blockDim.x + threadIdx.x;           // (KP/32)*octiles*512 == KP*C
    int total = (KP >> 5) * octiles * 512;
    if (idx >= total) return;
    int e    = idx & 15;
    int lane = (idx >> 4) & 31;
    int tile = idx >> 9;                                       // chunk*octiles + oc
    int oc    = tile % octiles;
    int chunk = tile / octiles;
    int hf = lane >> 4, l16 = lane & 15;
    int k = (chunk << 5) + (hf << 4) + e;
    int o = (oc << 4) + l16;
    float v = 0.f;
    if (k < 27 * C) {
        int tap = k / C;
        int ci  = k - tap * C;
        v = w[(o * C + ci) * 27 + tap];
    }
    wS[idx] = (h16)v;
}

// ====================== splat (+upsample-add) =======================
// one wave per grid point: lanes=atoms compute Lorentzian, then lanes=channels.
// writes X (f32, channel-major [B,C,n^3]) and Xh (f16, channels-last [B,n^3,C])
__global__ void k_splat(const float* __restrict__ pos, const float* __restrict__ feat2,
                        const float* __restrict__ cb, const float* __restrict__ scale, int sidx,
                        const float* __restrict__ prev, int C, int n,
                        float* __restrict__ X, h16* __restrict__ Xh) {
    __shared__ float ls[8][NATOM];
    int lane = threadIdx.x & 31;
    int warp = threadIdx.x >> 5;
    long gp = (long)blockIdx.x * 8 + warp;                     // BATCH*n^3 total, multiple of 8
    int n3 = n * n * n;
    int b = (int)(gp / n3);
    int g = (int)(gp - (long)b * n3);
    int i = g / (n * n), j = (g / n) % n, kk = g % n;
    float step = LBOX / (float)(n - 1);
    float gx = i * step, gy = j * step, gz = kk * step;
    const float* p = &pos[(b * NATOM + lane) * 3];
    float dx = gx - p[0], dy = gy - p[1], dz = gz - p[2];
    float d2 = fmaxf(dx * dx + dy * dy + dz * dz, 0.f);
    float sc = scale[sidx];
    ls[warp][lane] = sc / (sc * sc + d2);
    __syncthreads();

    // half-pixel trilinear source coords (for s>0 upsample)
    int h2 = n >> 1;
    float si = (i + 0.5f) * 0.5f - 0.5f;
    float sj = (j + 0.5f) * 0.5f - 0.5f;
    float sk = (kk + 0.5f) * 0.5f - 0.5f;
    int i0 = (int)floorf(si), j0 = (int)floorf(sj), k0 = (int)floorf(sk);
    float fi = si - i0, fj = sj - j0, fk = sk - k0;
    int i1 = i0 + 1, j1 = j0 + 1, k1 = k0 + 1;
    i0 = i0 < 0 ? 0 : (i0 >= h2 ? h2 - 1 : i0);
    i1 = i1 < 0 ? 0 : (i1 >= h2 ? h2 - 1 : i1);
    j0 = j0 < 0 ? 0 : (j0 >= h2 ? h2 - 1 : j0);
    j1 = j1 < 0 ? 0 : (j1 >= h2 ? h2 - 1 : j1);
    k0 = k0 < 0 ? 0 : (k0 >= h2 ? h2 - 1 : k0);
    k1 = k1 < 0 ? 0 : (k1 >= h2 ? h2 - 1 : k1);
    int h22 = h2 * h2;

    for (int o = lane; o < C; o += 32) {
        float acc = cb[o];
        const float* f2 = &feat2[(b * NATOM) * C + o];
#pragma unroll 8
        for (int a = 0; a < NATOM; ++a) acc += ls[warp][a] * f2[a * C];
        if (prev) {
            const float* pv = prev + ((size_t)b * C + o) * (size_t)(h2 * h22);
            float c00 = pv[i0 * h22 + j0 * h2 + k0] * (1.f - fk) + pv[i0 * h22 + j0 * h2 + k1] * fk;
            float c01 = pv[i0 * h22 + j1 * h2 + k0] * (1.f - fk) + pv[i0 * h22 + j1 * h2 + k1] * fk;
            float c10 = pv[i1 * h22 + j0 * h2 + k0] * (1.f - fk) + pv[i1 * h22 + j0 * h2 + k1] * fk;
            float c11 = pv[i1 * h22 + j1 * h2 + k0] * (1.f - fk) + pv[i1 * h22 + j1 * h2 + k1] * fk;
            float c0 = c00 * (1.f - fj) + c01 * fj;
            float c1 = c10 * (1.f - fj) + c11 * fj;
            acc += c0 * (1.f - fi) + c1 * fi;
        }
        X[((size_t)b * C + o) * (size_t)n3 + g] = acc;         // channel-major f32
        Xh[((size_t)b * n3 + g) * (size_t)C + o] = (h16)acc;   // channels-last f16
    }
}

// ================= 3x3x3 circular conv, implicit GEMM ===============
// one wave: 16-wide x-strip (M) x 16 output channels (N); K = 27*C padded to KP.
// Activations f16 channels-last [B,n,n,n,C]: each lane's A fragment = two
// contiguous 8xf16 runs -> 2x global_load_b128 per chunk. Weights pre-swizzled
// into B-fragment order -> one contiguous 32B load per lane per chunk.
// mode 0: outh = relu(conv + bias)             (f16, channels-last)
// mode 1: outf = relu(resid + conv + bias)     (f32, channel-major; in-place OK)
__global__ void k_conv3(const h16* __restrict__ inh, const h16* __restrict__ wS,
                        const float* __restrict__ bias, const float* __restrict__ resid,
                        float* __restrict__ outf, h16* __restrict__ outh,
                        int n, int C, int logC, int KP, int mode) {
    int lane = threadIdx.x & 31;
    int wid  = blockIdx.x * (blockDim.x >> 5) + (threadIdx.x >> 5);
    int strips  = n >> 4;
    int octiles = C >> 4;
    int t = wid;
    int oc  = t % octiles;  t /= octiles;
    int xs  = (t % strips) << 4;  t /= strips;
    int y   = t % n;  t /= n;
    int z   = t % n;  t /= n;
    int b   = t;
    int hf = lane >> 4, l16 = lane & 15;
    int Ktot = 27 * C, Cm1 = C - 1;
    int xm = xs + l16;                                         // this lane's output x (= A row)
    v8f acc = {};
    for (int kc = 0; kc < KP; kc += 32) {
        // ---- B fragment: one contiguous 32-byte load per lane ----
        size_t boff = (((size_t)(kc >> 5) * octiles + oc) << 9) + ((size_t)lane << 4);
        v16h bf = *(const v16h*)(wS + boff);
        if (kc + 32 < KP)
            __builtin_prefetch(wS + boff + ((size_t)octiles << 9), 0, 3);  // next chunk
        // ---- A fragment: two contiguous 8xf16 runs per lane ----
        v16h a;
#pragma unroll
        for (int r = 0; r < 2; ++r) {
            int kb = kc + (hf << 4) + (r << 3);                // run base (aligned to 8)
            v8h rv = {};
            if (kb < Ktot) {
                int tap = kb >> logC;
                int cb0 = kb & Cm1;                            // 8 consecutive channels
                int dz = tap / 9, dy = (tap / 3) % 3, dxp = tap % 3;
                int zz = z + dz - 1;   zz += (zz < 0) ? n : 0;  zz -= (zz >= n) ? n : 0;
                int yy = y + dy - 1;   yy += (yy < 0) ? n : 0;  yy -= (yy >= n) ? n : 0;
                int xx = xm + dxp - 1; xx += (xx < 0) ? n : 0;  xx -= (xx >= n) ? n : 0;
                rv = *(const v8h*)(inh + ((((size_t)b * n + zz) * n + yy) * n + xx) * C + cb0);
            }
#pragma unroll
            for (int e = 0; e < 8; ++e) a[(r << 3) + e] = rv[e];
        }
        acc = __builtin_amdgcn_wmma_f32_16x16x32_f16(false, a, false, bf, (short)0, acc, false, false);
    }
    int o = (oc << 4) + l16;
    float bv = bias[o];
#pragma unroll
    for (int v = 0; v < 8; ++v) {
        int x = xs + v + (hf << 3);                            // D layout: M = v (+8 for hi half)
        float val = acc[v] + bv;
        if (mode == 0) {
            val = fmaxf(val, 0.f);
            size_t idx = ((((size_t)b * n + z) * n + y) * n + x) * (size_t)C + o;
            outh[idx] = (h16)val;
        } else {
            size_t idx = (((size_t)b * C + o) * n + z) * (size_t)n * n + (size_t)y * n + x;
            val = fmaxf(resid[idx] + val, 0.f);
            outf[idx] = val;
        }
    }
}

// ====================== 1x1x1 channel reduce ========================
__global__ void k_outconv(const float* __restrict__ X, const float* __restrict__ ow,
                          const float* __restrict__ ob, int C, int Cn, int n3,
                          float* __restrict__ dst) {
    long idx = (long)blockIdx.x * blockDim.x + threadIdx.x;    // BATCH*Cn*n3
    if (idx >= (long)BATCH * Cn * n3) return;
    int g = (int)(idx % n3);
    long t = idx / n3;
    int o = (int)(t % Cn);
    int b = (int)(t / Cn);
    float acc = ob[o];
    for (int c = 0; c < C; ++c) acc += ow[o * C + c] * X[((size_t)b * C + c) * (size_t)n3 + g];
    dst[idx] = acc;
}

// ============================== host ================================

extern "C" void kernel_launch(void* const* d_in, const int* in_sizes, int n_in,
                              void* d_out, int out_size, void* d_ws, size_t ws_size,
                              hipStream_t stream) {
    const float* pos    = (const float*)d_in[0];
    const float* embed  = (const float*)d_in[1];
    const float* wpos   = (const float*)d_in[2];
    const float* bpos   = (const float*)d_in[3];
    const float* wmsg   = (const float*)d_in[4];
    const float* wupd   = (const float*)d_in[5];
    const float* scale  = (const float*)d_in[6];
    const int*   classes = (const int*)d_in[37];
    const int*   edges   = (const int*)d_in[38];
    int nedge = in_sizes[38] / 2;
    float* out = (float*)d_out;

    char* ws = (char*)d_ws;
    size_t off = 0;
    auto wsa = [&](size_t bytes) -> char* {
        char* p = ws + off;
        off += (bytes + 255) & ~(size_t)255;
        return p;
    };
    float* hbuf  = (float*)wsa((size_t)TOT * EMB * 4);
    float* agg   = (float*)wsa((size_t)TOT * EMB * 4);
    float* feat  = (float*)wsa((size_t)TOT * 32 * 4);
    float* feat2 = (float*)wsa((size_t)TOT * 64 * 4);
    float* bufX  = (float*)wsa((size_t)16777216 * 4);   // max B*C*n^3 (stage2: 4*16*64^3)
    float* bufP  = (float*)wsa((size_t)2097152 * 4);    // max inter-stage output (4*16*32^3)
    h16*   Xh    = (h16*)  wsa((size_t)16777216 * 2);
    h16*   Hh    = (h16*)  wsa((size_t)16777216 * 2);
    h16*   wS1   = (h16*)  wsa((size_t)1728 * 64 * 2);
    h16*   wS2   = (h16*)  wsa((size_t)1728 * 64 * 2);

    // ---- MPNN ----
    k_prep_h<<<(TOT * EMB + 255) / 256, 256, 0, stream>>>(pos, embed, wpos, bpos, classes, hbuf);
    for (int r = 0; r < 3; ++r) {
        (void)hipMemsetAsync(agg, 0, (size_t)TOT * EMB * 4, stream);
        k_msg<<<nedge, EMB, 0, stream>>>(hbuf, wmsg, edges, nedge, agg);
        k_update_wmma<<<8, 256, 0, stream>>>(agg, wupd, hbuf);
    }

    // ---- stages ----
    const int CCs[4] = {64, 32, 16, 1};
    const int PCs[3] = {32, 32, 16};
    const int NSs[3] = {16, 32, 64};
    for (int s = 0; s < 3; ++s) {
        const float* pw  = (const float*)d_in[7 + s * 10 + 0];
        const float* pb  = (const float*)d_in[7 + s * 10 + 1];
        const float* cw  = (const float*)d_in[7 + s * 10 + 2];
        const float* cb  = (const float*)d_in[7 + s * 10 + 3];
        const float* r1w = (const float*)d_in[7 + s * 10 + 4];
        const float* r1b = (const float*)d_in[7 + s * 10 + 5];
        const float* r2w = (const float*)d_in[7 + s * 10 + 6];
        const float* r2b = (const float*)d_in[7 + s * 10 + 7];
        const float* oww = (const float*)d_in[7 + s * 10 + 8];
        const float* obb = (const float*)d_in[7 + s * 10 + 9];
        int C = CCs[s], PC = PCs[s], n = NSs[s];
        int n3 = n * n * n;
        int KP = ((27 * C + 31) / 32) * 32;
        int octiles = C / 16;
        int logC = (C == 16) ? 4 : ((C == 32) ? 5 : 6);

        k_feat<<<(TOT * PC + 255) / 256, 256, 0, stream>>>(hbuf, pw, pb, PC, feat);
        k_feat2<<<(TOT * C + 255) / 256, 256, 0, stream>>>(feat, cw, PC, C, feat2);
        k_wswz<<<(KP * C + 255) / 256, 256, 0, stream>>>(r1w, C, KP, octiles, wS1);
        k_wswz<<<(KP * C + 255) / 256, 256, 0, stream>>>(r2w, C, KP, octiles, wS2);

        k_splat<<<BATCH * n3 / 8, 256, 0, stream>>>(pos, feat2, cb, scale, s,
                                                    (s == 0) ? nullptr : bufP,
                                                    C, n, bufX, Xh);

        int waves = BATCH * n * n * (n / 16) * (C / 16);
        k_conv3<<<waves / 8, 256, 0, stream>>>(Xh, wS1, r1b, nullptr, nullptr, Hh,
                                               n, C, logC, KP, 0);
        k_conv3<<<waves / 8, 256, 0, stream>>>(Hh, wS2, r2b, bufX, bufX, nullptr,
                                               n, C, logC, KP, 1);

        int Cn = CCs[s + 1];
        float* dst = (s == 2) ? out : bufP;
        long tot = (long)BATCH * Cn * n3;
        k_outconv<<<(int)((tot + 255) / 256), 256, 0, stream>>>(bufX, oww, obb, C, Cn, n3, dst);
    }
}
